// AVWGCN_48292612276865
// MI455X (gfx1250) — matrix operand order repaired and checked by reading, not compiled, not run on hardware
//
#include <hip/hip_runtime.h>

typedef _Float16 h16 __attribute__((ext_vector_type(16)));
typedef _Float16 h8  __attribute__((ext_vector_type(8)));
typedef float    f8  __attribute__((ext_vector_type(8)));
typedef float    f4  __attribute__((ext_vector_type(4)));

#define NNODE 2048
#define NB    64
#define CC    64
#define NEMB  16
#define BC    4096   // NB*CC
#define KD    192    // 3*CC

__device__ __forceinline__ f8 wmma_f16(h16 a, h16 b, f8 c) {
  return __builtin_amdgcn_wmma_f32_16x16x32_f16(
      false, a, false, b, (short)0, c, false, false);
}

// CDNA5 async copy: 16B global -> LDS per lane, tracked by ASYNCcnt.
// GVS mode: mem_addr = SADDR(u64) + VADDR(u32 byte offset); VDST = LDS byte addr.
__device__ __forceinline__ void async_copy16(unsigned lds_addr, unsigned goff,
                                             const void* gbase) {
  asm volatile("global_load_async_to_lds_b128 %0, %1, %2"
               :: "v"(lds_addr), "v"(goff), "s"(gbase) : "memory");
}
__device__ __forceinline__ void wait_async() {
  asm volatile("s_wait_asynccnt 0x0" ::: "memory");
}

// ---------------------------------------------------------------------------
// K0a: x [B,N,C] f32 -> X0T [BC, N] f16   (X0T[b*64+c][n] = x[b,n,c])
// ---------------------------------------------------------------------------
__global__ __launch_bounds__(256) void k_x_to_T(const float* __restrict__ x,
                                                _Float16* __restrict__ X0T) {
  __shared__ _Float16 tile[64 * 80];            // [n_local][c], padded
  const int n0 = blockIdx.x * 64;
  const int b  = blockIdx.y;
  const int tid = threadIdx.x;
#pragma unroll
  for (int it = 0; it < 4; ++it) {
    int t = tid + it * 256;                     // 0..1023
    int r = t >> 4;                             // n_local
    int c4 = (t & 15) * 4;
    f4 v = *(const f4*)&x[((size_t)b * NNODE + n0 + r) * CC + c4];
#pragma unroll
    for (int q = 0; q < 4; ++q) tile[r * 80 + c4 + q] = (_Float16)v[q];
  }
  __syncthreads();
#pragma unroll
  for (int it = 0; it < 2; ++it) {
    int t = tid + it * 256;                     // 0..511
    int c = t >> 3;
    int nn0 = (t & 7) * 8;
    h8 o;
#pragma unroll
    for (int v = 0; v < 8; ++v) o[v] = tile[(nn0 + v) * 80 + c];
    __builtin_nontemporal_store(o, (h8*)&X0T[(size_t)(b * CC + c) * NNODE + n0 + nn0]);
  }
}

// ---------------------------------------------------------------------------
// K0b: x [B,N,C] f32 -> Xt [N, BC] f16   (Xt[n][b*64+c] = x[b,n,c])
// ---------------------------------------------------------------------------
__global__ __launch_bounds__(256) void k_x_to_node(const float* __restrict__ x,
                                                   _Float16* __restrict__ Xt) {
  int idx = blockIdx.x * 256 + threadIdx.x;
  int n = idx >> 12;
  int r = idx & 4095;
  int b = r >> 6;
  int c = r & 63;
  Xt[idx] = (_Float16)x[(size_t)b * (NNODE * CC) + (size_t)n * CC + c];
}

// ---------------------------------------------------------------------------
// K1: adjacency  A = softmax(relu(E E^T)) rows -> f16 [N,N]
// ---------------------------------------------------------------------------
__global__ __launch_bounds__(256) void k_adjacency(const float* __restrict__ E,
                                                   _Float16* __restrict__ A) {
  __shared__ float row[NNODE];
  __shared__ float red[256];
  __shared__ float en[NEMB];
  const int n = blockIdx.x;
  const int tid = threadIdx.x;

  if (tid < NEMB) en[tid] = E[n * NEMB + tid];
  __syncthreads();

  float lmax = 0.0f;
  for (int m = tid; m < NNODE; m += 256) {
    const float* em = &E[m * NEMB];
    float d = 0.0f;
#pragma unroll
    for (int e = 0; e < NEMB; ++e) d += en[e] * em[e];
    d = fmaxf(d, 0.0f);
    row[m] = d;
    lmax = fmaxf(lmax, d);
  }
  red[tid] = lmax;
  __syncthreads();
  for (int s = 128; s > 0; s >>= 1) {
    if (tid < s) red[tid] = fmaxf(red[tid], red[tid + s]);
    __syncthreads();
  }
  const float rmax = red[0];
  __syncthreads();

  float lsum = 0.0f;
  for (int m = tid; m < NNODE; m += 256) {
    float e = __expf(row[m] - rmax);
    row[m] = e;
    lsum += e;
  }
  red[tid] = lsum;
  __syncthreads();
  for (int s = 128; s > 0; s >>= 1) {
    if (tid < s) red[tid] += red[tid + s];
    __syncthreads();
  }
  const float inv = 1.0f / red[0];

  for (int m = tid; m < NNODE; m += 256)
    A[(size_t)n * NNODE + m] = (_Float16)(row[m] * inv);
}

// ---------------------------------------------------------------------------
// K2: per-node weights, K-transposed:  Wt[n][o][k*64+i] = sum_d E[n,d]*Wp[d,k,i,o]
// ---------------------------------------------------------------------------
__global__ __launch_bounds__(256) void k_make_weights(const float* __restrict__ E,
                                                      const float* __restrict__ Wp,
                                                      _Float16* __restrict__ Wt) {
  int idx = blockIdx.x * 256 + threadIdx.x;
  int n   = idx / (CC * KD);
  int rem = idx - n * (CC * KD);
  int o   = rem / KD;
  int ki  = rem - o * KD;
  int k   = ki >> 6;
  int i   = ki & 63;
  const float* e = &E[n * NEMB];
  float s = 0.0f;
#pragma unroll
  for (int d = 0; d < NEMB; ++d)
    s += e[d] * Wp[d * (3 * CC * CC) + k * (CC * CC) + i * CC + o];
  __builtin_nontemporal_store((_Float16)s, &Wt[idx]);
}

// ---------------------------------------------------------------------------
// K3: WMMA GEMM, transposed-output form, double-buffered async LDS staging.
//   OutT[j][m] = f16( scale * sum_k Adj[m][k] * BT[j][k]  (+ subscale*SubT[j][m]) )
// Block tile 256(m) x 128(j), 8 waves = 4m x 2j, wave = 64x64 = 4x4 WMMA tiles.
// Tile i+1 streams in via ASYNCcnt while tile i runs 16 WMMAs.
// ---------------------------------------------------------------------------
template <bool SUB>
__global__ __launch_bounds__(256) void k_gemm(const _Float16* __restrict__ Adj,
                                              const _Float16* __restrict__ BT,
                                              const _Float16* __restrict__ SubT,
                                              _Float16* __restrict__ OutT,
                                              float scale, float subscale) {
  __shared__ _Float16 As[2][256 * 32];   // [m][k], double buffered
  __shared__ _Float16 Bs[2][128 * 32];   // [j][k], double buffered

  const int tid  = threadIdx.x;
  const int lane = tid & 31;
  const int wave = tid >> 5;
  const int wm   = wave & 3;             // m strip of 64
  const int wj   = wave >> 2;            // j strip of 64
  const int gm0  = blockIdx.y * 256;
  const int gj0  = blockIdx.x * 128;
  const int l15  = lane & 15;
  const int lhi  = lane >> 4;

  auto stage = [&](int kk, int bi) {
#pragma unroll
    for (int it = 0; it < 4; ++it) {     // A: 256x32 halves = 1024 x 16B
      int t   = tid + it * 256;
      int row = t >> 2;
      int col = (t & 3) * 8;
      async_copy16((unsigned)(size_t)&As[bi][row * 32 + col],
                   (unsigned)(((gm0 + row) * NNODE + kk + col) * 2), Adj);
    }
#pragma unroll
    for (int it = 0; it < 2; ++it) {     // B: 128x32 halves = 512 x 16B
      int t   = tid + it * 256;
      int row = t >> 2;
      int col = (t & 3) * 8;
      async_copy16((unsigned)(size_t)&Bs[bi][row * 32 + col],
                   (unsigned)(((gj0 + row) * NNODE + kk + col) * 2), BT);
    }
  };

  stage(0, 0);
  wait_async();
  __syncthreads();

  f8 acc[4][4] = {};

  for (int kk = 0; kk < NNODE; kk += 32) {
    const int bi = (kk >> 5) & 1;
    if (kk + 32 < NNODE) stage(kk + 32, bi ^ 1);   // overlap DMA with WMMA

    // B fragments (32x16): lane = column j, 16 contiguous K halves; reused 4x
    h16 bf[4];
#pragma unroll
    for (int js = 0; js < 4; ++js) {
      int jcol = wj * 64 + js * 16 + l15;
      int ko   = lhi * 16;
      h16 b;
      *(h8*)&b       = *(const h8*)&Bs[bi][jcol * 32 + ko];
      *((h8*)&b + 1) = *(const h8*)&Bs[bi][jcol * 32 + ko + 8];
      bf[js] = b;
    }
    // A fragments (16x32): lane<16 K{0..15}, lane>=16 K{16..31}
#pragma unroll
    for (int ms = 0; ms < 4; ++ms) {
      int mrow = wm * 64 + ms * 16 + l15;
      int ko   = lhi * 16;
      h16 a;
      *(h8*)&a       = *(const h8*)&As[bi][mrow * 32 + ko];
      *((h8*)&a + 1) = *(const h8*)&As[bi][mrow * 32 + ko + 8];
#pragma unroll
      for (int js = 0; js < 4; ++js)
        acc[ms][js] = wmma_f16(a, bf[js], acc[ms][js]);
    }

    wait_async();
    __syncthreads();
  }

  // epilogue: lane holds col j fixed, 8 consecutive m -> contiguous in OutT[j][*]
#pragma unroll
  for (int ms = 0; ms < 4; ++ms) {
#pragma unroll
    for (int js = 0; js < 4; ++js) {
      int j  = gj0 + wj * 64 + js * 16 + l15;
      int m0 = gm0 + wm * 64 + ms * 16 + lhi * 8;
      h8 o;
      if (SUB) {
        h8 s = *(const h8*)&SubT[(size_t)j * NNODE + m0];
#pragma unroll
        for (int v = 0; v < 8; ++v)
          o[v] = (_Float16)(scale * acc[ms][js][v] + subscale * (float)s[v]);
      } else {
#pragma unroll
        for (int v = 0; v < 8; ++v) o[v] = (_Float16)(scale * acc[ms][js][v]);
      }
      __builtin_nontemporal_store(o, (h8*)&OutT[(size_t)j * NNODE + m0]);
    }
  }
}

// ---------------------------------------------------------------------------
// K3b: tiled transpose  In [BC, N] f16 -> Out [N, BC] f16 (coalesced both sides)
// ---------------------------------------------------------------------------
__global__ __launch_bounds__(256) void k_transpose_TN(const _Float16* __restrict__ In,
                                                      _Float16* __restrict__ Out) {
  __shared__ _Float16 tile[64 * 80];            // [j_local][n_local], padded
  const int n0 = blockIdx.x * 64;
  const int j0 = blockIdx.y * 64;
  const int tid = threadIdx.x;
#pragma unroll
  for (int it = 0; it < 2; ++it) {
    int t  = tid + it * 256;                    // 0..511
    int r  = t >> 3;                            // j_local
    int c0 = (t & 7) * 8;                       // n_local
    *(h8*)&tile[r * 80 + c0] = *(const h8*)&In[(size_t)(j0 + r) * NNODE + n0 + c0];
  }
  __syncthreads();
#pragma unroll
  for (int it = 0; it < 2; ++it) {
    int t   = tid + it * 256;
    int rr  = t >> 3;                           // n_local
    int cc0 = (t & 7) * 8;                      // j_local
    h8 o;
#pragma unroll
    for (int v = 0; v < 8; ++v) o[v] = tile[(cc0 + v) * 80 + rr];
    __builtin_nontemporal_store(o, (h8*)&Out[(size_t)(n0 + rr) * BC + j0 + cc0]);
  }
}

// ---------------------------------------------------------------------------
// K4: per-node GEMM, transposed orientation (M = C_out, N = batch):
//   out[b,n,o] = sum_ki Wt[n][o][ki] * xg[n][b][ki] + bias[n][o]
// 1 block = node, 4 waves = o-strips of 16, 4 b-tiles/wave, K=192 = 6 steps.
// D frag: lane = batch col, 8 consecutive o -> two float4 stores.
// ---------------------------------------------------------------------------
__global__ __launch_bounds__(128) void k_output(const _Float16* __restrict__ Xt,
                                                const _Float16* __restrict__ Y1,
                                                const _Float16* __restrict__ Xg2,
                                                const _Float16* __restrict__ Wt,
                                                const float* __restrict__ E,
                                                const float* __restrict__ bp,
                                                float* __restrict__ Out) {
  __shared__ float bias_sh[CC];
  const int n    = blockIdx.x;
  const int lane = threadIdx.x & 31;
  const int wave = threadIdx.x >> 5;            // o-strip
  const int l15  = lane & 15;
  const int lhi  = lane >> 4;

  if (threadIdx.x < CC) {
    float s = 0.0f;
#pragma unroll
    for (int d = 0; d < NEMB; ++d) s += E[n * NEMB + d] * bp[d * CC + threadIdx.x];
    bias_sh[threadIdx.x] = s;
  }
  __syncthreads();

  const _Float16* srcs[3] = {Xt + (size_t)n * BC, Y1 + (size_t)n * BC,
                             Xg2 + (size_t)n * BC};
  const _Float16* wn = Wt + (size_t)n * (CC * KD);

  f8 acc[4] = {};

#pragma unroll
  for (int step = 0; step < 6; ++step) {
    // A fragment from Wt: rows = o, K = step*32 slice of 192
    int orow = wave * 16 + l15;
    int ko   = step * 32 + lhi * 16;
    h16 a;
    *(h8*)&a       = *(const h8*)&wn[orow * KD + ko];
    *((h8*)&a + 1) = *(const h8*)&wn[orow * KD + ko + 8];
    const _Float16* src = srcs[step >> 1];
#pragma unroll
    for (int t = 0; t < 4; ++t) {
      // B fragment: lane = batch col, 16 contiguous ki within 64-wide source
      int bcol = t * 16 + l15;
      int kb   = (step & 1) * 32 + lhi * 16;
      h16 b;
      *(h8*)&b       = *(const h8*)&src[bcol * CC + kb];
      *((h8*)&b + 1) = *(const h8*)&src[bcol * CC + kb + 8];
      acc[t] = wmma_f16(a, b, acc[t]);
    }
  }

#pragma unroll
  for (int t = 0; t < 4; ++t) {
    int b  = t * 16 + l15;
    int o0 = wave * 16 + lhi * 8;
    float* dst = &Out[((size_t)b * NNODE + n) * CC + o0];
    f4 lo, hi;
#pragma unroll
    for (int v = 0; v < 4; ++v) lo[v] = acc[t][v] + bias_sh[o0 + v];
#pragma unroll
    for (int v = 0; v < 4; ++v) hi[v] = acc[t][v + 4] + bias_sh[o0 + 4 + v];
    *(f4*)dst       = lo;
    *(f4*)(dst + 4) = hi;
  }
}

// ---------------------------------------------------------------------------
// Launch
// ---------------------------------------------------------------------------
extern "C" void kernel_launch(void* const* d_in, const int* in_sizes, int n_in,
                              void* d_out, int out_size, void* d_ws, size_t ws_size,
                              hipStream_t stream) {
  const float* x  = (const float*)d_in[0];   // [64,2048,64]
  const float* E  = (const float*)d_in[1];   // [2048,16]
  const float* Wp = (const float*)d_in[2];   // [16,3,64,64]
  const float* bp = (const float*)d_in[3];   // [16,64]
  float* out = (float*)d_out;                // [64,2048,64]

  char* ws = (char*)d_ws;
  _Float16* Adj  = (_Float16*)(ws);                          //  8 MB [N,N]
  _Float16* X0T  = (_Float16*)(ws + ((size_t)8   << 20));    // 16 MB [BC,N]
  _Float16* Xt   = (_Float16*)(ws + ((size_t)24  << 20));    // 16 MB [N,BC]
  _Float16* Y1T  = (_Float16*)(ws + ((size_t)40  << 20));    // 16 MB [BC,N]
  _Float16* Y1   = (_Float16*)(ws + ((size_t)56  << 20));    // 16 MB [N,BC]
  _Float16* Xg2T = (_Float16*)(ws + ((size_t)72  << 20));    // 16 MB [BC,N]
  _Float16* Xg2  = (_Float16*)(ws + ((size_t)88  << 20));    // 16 MB [N,BC]
  _Float16* Wt   = (_Float16*)(ws + ((size_t)104 << 20));    // 48 MB [N,64,192]

  k_x_to_T<<<dim3(NNODE / 64, NB), 256, 0, stream>>>(x, X0T);
  k_x_to_node<<<(NNODE * BC) / 256, 256, 0, stream>>>(x, Xt);
  k_adjacency<<<NNODE, 256, 0, stream>>>(E, Adj);
  k_make_weights<<<(NNODE * CC * KD) / 256, 256, 0, stream>>>(E, Wp, Wt);
  // Y1T[j][m] = (A @ X)^T
  k_gemm<false><<<dim3(BC / 128, NNODE / 256), 256, 0, stream>>>(
      Adj, X0T, (const _Float16*)nullptr, Y1T, 1.0f, 0.0f);
  k_transpose_TN<<<dim3(NNODE / 64, BC / 64), 256, 0, stream>>>(Y1T, Y1);
  // Xg2T = (2*A@Y1 - X)^T  == (T2 @ x)^T
  k_gemm<true><<<dim3(BC / 128, NNODE / 256), 256, 0, stream>>>(
      Adj, Y1T, X0T, Xg2T, 2.0f, -1.0f);
  k_transpose_TN<<<dim3(NNODE / 64, BC / 64), 256, 0, stream>>>(Xg2T, Xg2);
  k_output<<<NNODE, 128, 0, stream>>>(Xt, Y1, Xg2, Wt, E, bp, out);
}
